// SeqPinn_58428735095531
// MI455X (gfx1250) — matrix operand order repaired and checked by reading, not compiled, not run on hardware
//
#include <hip/hip_runtime.h>
#include <cstddef>
#include <cstdint>

// ---------------------------------------------------------------------------
// SeqPinn GRU seq2seq on gfx1250 (MI455X).  bf16 WMMA GEMMs, f32 accumulate,
// fused gate epilogues, weights pre-swizzled into WMMA B-fragment layout.
// All hot fragment loads forced to global address space (global_load_b128,
// LOADcnt-only) instead of flat (which also ties up DScnt).
// ---------------------------------------------------------------------------

typedef __attribute__((ext_vector_type(16))) __bf16 v16bf;
typedef __attribute__((ext_vector_type(8)))  __bf16 v8bf;
typedef __attribute__((ext_vector_type(8)))  float  v8f;

#define HSZ   256            // GRU hidden
#define BSZ   256            // batch
#define TSTP  97             // timesteps
#define ENCO  48
#define T8    (TSTP * 8)     // input_X row stride (8 features)
#define EW    192            // concat(oE,oI,oH) width
#define HELEM (HSZ * BSZ)    // 65536 elements per branch state

#define GLOBAL_AS __attribute__((address_space(1)))

// ---------------------------- device helpers -------------------------------

__device__ __forceinline__ v8f wmma_bf16(v16bf a, v16bf b, v8f c) {
  // D = A(16xK=32) * B(32x16) + C, f32 accumulate
  return __builtin_amdgcn_wmma_f32_16x16x32_bf16(
      false, a, false, b, (short)0, c, false, false);
}

// Force a load through the global (AS1) path: global_load_b128, LOADcnt only.
__device__ __forceinline__ v16bf gload16(const __bf16* p) {
  return *(GLOBAL_AS const v16bf*)(uintptr_t)p;
}
__device__ __forceinline__ v8bf gload8(const __bf16* p) {
  return *(GLOBAL_AS const v8bf*)(uintptr_t)p;
}

// A-fragment (16x32 bf16) from a row-major bf16 matrix (ld elements/row).
// ISA layout: lane holds row M = lane&15; element e: K = (e>=8)*16 + half*8 + (e&7)
__device__ __forceinline__ v16bf load_a_frag(const __bf16* base, int row,
                                             int ld, int kbase, int half) {
  const __bf16* p = base + row * ld + kbase + half * 8;
  v8bf lo = gload8(p);        // K = kbase + half*8 .. +7
  v8bf hi = gload8(p + 16);   // K = kbase + 16 + half*8 .. +7
  return __builtin_shufflevector(lo, hi, 0, 1, 2, 3, 4, 5, 6, 7,
                                 8, 9, 10, 11, 12, 13, 14, 15);
}

// B-fragment from pre-swizzled weights: frag[((nTile*ksteps + ks)*32 + lane)*16]
__device__ __forceinline__ v16bf load_b_frag(const __bf16* frag, int nTile,
                                             int ksteps, int ks, int lane) {
  return gload16(frag + (((nTile * ksteps + ks) << 5) + lane) * 16);
}

__device__ __forceinline__ float sigmoidf_(float x) {
  return 1.0f / (1.0f + expf(-x));
}

// ---------------------------- kernarg structs ------------------------------

struct BranchK {
  const float* Wih;            // (768, in_n) f32
  const float* bih;            // (768)
  const float* bhh;            // (768)
  const unsigned short* whhF;  // bf16 fragment-swizzled Whh^T (K=256, N=768)
  int in_n;                    // total input features (incl. zone)
  int use_zone;                // 1 -> feature 0 is the zone scalar
  int nx;                      // # of columns pulled from input_X
  int fidx[4];                 // which input_X columns
};
struct Branch3 { BranchK b[3]; };

struct Out3 {
  const unsigned short* woF[3]; // bf16 frag W^T (K=256, N=64)
  const float* bo[3];           // (64)
};

struct HeadK {
  const unsigned short* w1F;    // bf16 frag W1^T (K=192, N=256)
  const float* b1;              // (256)
  const float* W2;              // (256)
};

// ------------------------------- kernels -----------------------------------

// Swizzle f32 weight W (N rows, K cols, row-major, used as B = W^T) into
// bf16 WMMA B-fragment order. B element (K=k, N=n) = W[n*K + k].
// Fragment K mapping: k = ks*32 + (e>=8)*16 + half*8 + (e&7).
__global__ void prep_frag(const float* __restrict__ W,
                          unsigned short* __restrict__ dst, int K, int total) {
  int f = blockIdx.x * 256 + threadIdx.x;
  if (f >= total) return;
  int ksteps = K >> 5;
  int e    = f & 15;
  int lane = (f >> 4) & 31;
  int grp  = f >> 9;
  int ks   = grp % ksteps;
  int nT   = grp / ksteps;
  int n = (nT << 4) + (lane & 15);
  int k = (ks << 5) + ((e >> 3) << 4) + ((lane >> 4) << 3) + (e & 7);
  __bf16 v = (__bf16)W[n * K + k];
  dst[f] = *(const unsigned short*)&v;
}

__global__ void k_init(float* __restrict__ hF, unsigned short* __restrict__ hBF,
                       float* __restrict__ zone, const float* __restrict__ X) {
  int idx = blockIdx.x * 256 + threadIdx.x;
  if (idx < 3 * HELEM) {
    hF[idx] = 0.0f;
    hBF[idx] = 0;
  }
  if (idx < BSZ) zone[idx] = X[idx * T8 + 0];   // input_X[:,0,0]
}

__global__ void k_zone_reset(float* __restrict__ zone,
                             const float* __restrict__ X) {
  int b = threadIdx.x;
  zone[b] = X[b * T8 + ENCO * 8 + 0];           // input_X[:,48,0]
}

// Fused GRU cell for all 3 branches.  grid = 48 WGs (3 branch x 4 rowblk x 4 colblk)
// WG tile: 64 batch rows x 64 hidden cols, computed for all 3 gates.
__global__ void __launch_bounds__(256)
k1_gru(Branch3 P,
       const unsigned short* __restrict__ hcurBF, const float* __restrict__ hcurF,
       float* __restrict__ hnxtF, unsigned short* __restrict__ hnxtBF,
       const float* __restrict__ zone, const float* __restrict__ X, int t) {
  int bx = blockIdx.x;
  int branch = bx >> 4;
  int sub = bx & 15;
  int rb = sub >> 2;
  int cb = sub & 3;
  const BranchK bp = P.b[branch];

  int tid  = threadIdx.x;
  int wave = tid >> 5;
  int lane = tid & 31;
  int half = lane >> 4;
  int l16  = lane & 15;

  int rowBase = rb * 64 + (wave >> 1) * 16;     // batch rows
  int colBase = cb * 64 + (wave & 1) * 32;      // hidden cols (2 tiles)

  const __bf16* hbf = ((const __bf16*)hcurBF) + branch * HELEM;
  const __bf16* wf  = (const __bf16*)bp.whhF;

  v8f zf = {0.f, 0.f, 0.f, 0.f, 0.f, 0.f, 0.f, 0.f};
  v8f acc[3][2];
#pragma unroll
  for (int g = 0; g < 3; ++g)
#pragma unroll
    for (int c = 0; c < 2; ++c) acc[g][c] = zf;

  for (int ks = 0; ks < 8; ++ks) {              // K = 256 in steps of 32
    v16bf a = load_a_frag(hbf, rowBase + l16, HSZ, ks << 5, half);
#pragma unroll
    for (int g = 0; g < 3; ++g) {
#pragma unroll
      for (int ct = 0; ct < 2; ++ct) {
        int nT = g * 16 + (colBase >> 4) + ct;  // column tile in 768-wide gate dim
        v16bf b = load_b_frag(wf, nT, 8, ks, lane);
        acc[g][ct] = wmma_bf16(a, b, acc[g][ct]);
      }
    }
  }

  // Epilogue: tiny x@Wih^T + gates + state update.
#pragma unroll
  for (int v = 0; v < 8; ++v) {
    int row = rowBase + v + (half << 3);        // batch row for this C/D slot
    float xf[5];
    {
      int c = 0;
      if (bp.use_zone) xf[c++] = zone[row];
      const float* xr = X + row * T8 + t * 8;
      for (int q = 0; q < bp.nx; ++q) xf[c++] = xr[bp.fidx[q]];
    }
#pragma unroll
    for (int ct = 0; ct < 2; ++ct) {
      int j = colBase + (ct << 4) + l16;        // hidden unit index
      float ghR = acc[0][ct][v] + bp.bhh[j];
      float ghZ = acc[1][ct][v] + bp.bhh[256 + j];
      float ghN = acc[2][ct][v] + bp.bhh[512 + j];
      float giR = bp.bih[j];
      float giZ = bp.bih[256 + j];
      float giN = bp.bih[512 + j];
      const float* wr = bp.Wih + (size_t)j * bp.in_n;
      const float* wz = bp.Wih + (size_t)(256 + j) * bp.in_n;
      const float* wn = bp.Wih + (size_t)(512 + j) * bp.in_n;
      for (int k = 0; k < bp.in_n; ++k) {
        giR = fmaf(xf[k], wr[k], giR);
        giZ = fmaf(xf[k], wz[k], giZ);
        giN = fmaf(xf[k], wn[k], giN);
      }
      float r = sigmoidf_(giR + ghR);
      float z = sigmoidf_(giZ + ghZ);
      float n = tanhf(giN + r * ghN);
      size_t o = (size_t)branch * HELEM + (size_t)row * HSZ + j;
      float hold = hcurF[o];
      float hnew = (1.0f - z) * n + z * hold;
      hnxtF[o] = hnew;
      __bf16 hb = (__bf16)hnew;
      hnxtBF[o] = *(const unsigned short*)&hb;
    }
  }
}

// Branch outputs: o = h2 @ W^T + b -> e_bf16[:, branch*64 : +64]
// grid = 12 WGs (3 branch x 4 rowblk); WG tile 64 rows x 64 cols.
__global__ void __launch_bounds__(256)
k2_out(Out3 P, const unsigned short* __restrict__ h2BF,
       unsigned short* __restrict__ eBF) {
  int branch = blockIdx.x >> 2;
  int rb = blockIdx.x & 3;
  int tid  = threadIdx.x;
  int wave = tid >> 5;
  int lane = tid & 31;
  int half = lane >> 4;
  int l16  = lane & 15;
  int rowBase = rb * 64 + (wave >> 1) * 16;
  int colBase = (wave & 1) * 32;                // N = 64 -> 2 tiles per wave

  const __bf16* hbf = ((const __bf16*)h2BF) + branch * HELEM;
  const __bf16* wf  = (const __bf16*)P.woF[branch];
  const float*  bo  = P.bo[branch];

  v8f zf = {0.f, 0.f, 0.f, 0.f, 0.f, 0.f, 0.f, 0.f};
  v8f acc[2] = {zf, zf};
  for (int ks = 0; ks < 8; ++ks) {
    v16bf a = load_a_frag(hbf, rowBase + l16, HSZ, ks << 5, half);
#pragma unroll
    for (int ct = 0; ct < 2; ++ct) {
      int nT = (colBase >> 4) + ct;
      v16bf b = load_b_frag(wf, nT, 8, ks, lane);
      acc[ct] = wmma_bf16(a, b, acc[ct]);
    }
  }
#pragma unroll
  for (int v = 0; v < 8; ++v) {
    int row = rowBase + v + (half << 3);
#pragma unroll
    for (int ct = 0; ct < 2; ++ct) {
      int col = colBase + (ct << 4) + l16;
      float val = acc[ct][v] + bo[col];
      __bf16 hb = (__bf16)val;
      eBF[(size_t)row * EW + branch * 64 + col] = *(const unsigned short*)&hb;
    }
  }
}

// Head: relu(e @ W1^T + b1) . W2 -> zone += delta; out[b, t] = zone.
// grid = 16 WGs; WG: 16 rows x 256 cols, K = 192 (6 k-steps).
__global__ void __launch_bounds__(256)
k3_head(HeadK Hd, const unsigned short* __restrict__ eBF,
        float* __restrict__ zone, float* __restrict__ out, int t) {
  __shared__ float red[16][256];
  __shared__ float red2[16][16];

  int rb = blockIdx.x;
  int rowG0 = rb * 16;
  int tid  = threadIdx.x;
  int wave = tid >> 5;
  int lane = tid & 31;
  int half = lane >> 4;
  int l16  = lane & 15;
  int colBase = wave * 32;

  const __bf16* ebf = (const __bf16*)eBF;
  const __bf16* wf  = (const __bf16*)Hd.w1F;

  v8f zf = {0.f, 0.f, 0.f, 0.f, 0.f, 0.f, 0.f, 0.f};
  v8f acc[2] = {zf, zf};
  for (int ks = 0; ks < 6; ++ks) {
    v16bf a = load_a_frag(ebf, rowG0 + l16, EW, ks << 5, half);
#pragma unroll
    for (int ct = 0; ct < 2; ++ct) {
      int nT = wave * 2 + ct;
      v16bf b = load_b_frag(wf, nT, 6, ks, lane);
      acc[ct] = wmma_bf16(a, b, acc[ct]);
    }
  }
#pragma unroll
  for (int v = 0; v < 8; ++v) {
    int lrow = v + (half << 3);
#pragma unroll
    for (int ct = 0; ct < 2; ++ct) {
      int col = colBase + (ct << 4) + l16;
      float hv = acc[ct][v] + Hd.b1[col];
      hv = fmaxf(hv, 0.0f);
      red[lrow][col] = hv * Hd.W2[col];
    }
  }
  __syncthreads();
  {
    int row = tid >> 4, seg = tid & 15;
    float s = 0.0f;
#pragma unroll
    for (int c = 0; c < 16; ++c) s += red[row][seg * 16 + c];
    red2[row][seg] = s;
  }
  __syncthreads();
  if (tid < 16) {
    float d = 0.0f;
#pragma unroll
    for (int c = 0; c < 16; ++c) d += red2[tid][c];
    int gr = rowG0 + tid;
    float zn = zone[gr] + d;
    zone[gr] = zn;
    out[(size_t)gr * TSTP + t] = zn;
  }
}

// ------------------------------- host side ---------------------------------

extern "C" void kernel_launch(void* const* d_in, const int* in_sizes, int n_in,
                              void* d_out, int out_size, void* d_ws,
                              size_t ws_size, hipStream_t stream) {
  (void)out_size; (void)ws_size;
  // ---- locate inputs ----
  const float* X = nullptr;
  int pidx[64];
  int np = 0;
  for (int i = 0; i < n_in; ++i) {
    if (in_sizes[i] == BSZ * TSTP * 8 && !X) {
      X = (const float*)d_in[i];
    } else if (in_sizes[i] == 1) {
      // encoLen (fixed at 48 by the reference setup)
    } else if (np < 64) {
      pidx[np++] = i;
    }
  }
  if (!X || np < 42) return;

  struct Br { const float *Wih, *Whh, *bih, *bhh, *W, *b; };
  struct Hd { const float *W1, *b1, *W2; };
  Br br[2][3];  // [phase: 0=enc,1=dec][slot: 0=ext,1=int,2=hvac]
  Hd hd[2];

  int c = 0;
  auto nf = [&]() -> const float* { return (const float*)d_in[pidx[c++]]; };
  bool sortedMode = (in_sizes[pidx[0]] == 64 * HSZ);  // leading W => key-sorted
  if (sortedMode) {
    // dec_ext, dec_hvac, dec_int, dec_out, enc_ext, enc_hvac, enc_int, enc_out
    // branch group: W, b, Whh, Wih, bhh, bih ; head: W1, W2, b1
    const int phaseOrder[2] = {1, 0};
    const int slotOrder[3]  = {0, 2, 1};  // ext, hvac, int
    for (int pi = 0; pi < 2; ++pi) {
      int ph = phaseOrder[pi];
      for (int si = 0; si < 3; ++si) {
        Br& B = br[ph][slotOrder[si]];
        B.W = nf(); B.b = nf(); B.Whh = nf(); B.Wih = nf();
        B.bhh = nf(); B.bih = nf();
      }
      Hd& H = hd[ph];
      H.W1 = nf(); H.W2 = nf(); H.b1 = nf();
    }
  } else {
    // enc_ext, enc_int, enc_hvac, dec_ext, dec_int, dec_hvac, enc_out, dec_out
    // branch group: Wih, Whh, bih, bhh, W, b ; head: W1, b1, W2
    for (int ph = 0; ph < 2; ++ph)
      for (int s = 0; s < 3; ++s) {
        Br& B = br[ph][s];
        B.Wih = nf(); B.Whh = nf(); B.bih = nf(); B.bhh = nf();
        B.W = nf(); B.b = nf();
      }
    for (int ph = 0; ph < 2; ++ph) {
      Hd& H = hd[ph];
      H.W1 = nf(); H.b1 = nf(); H.W2 = nf();
    }
  }

  // ---- workspace layout ----
  char* wsb = (char*)d_ws;
  size_t cur = 0;
  auto alloc = [&](size_t bytes) -> char* {
    char* p = wsb + cur;
    cur = (cur + bytes + 255) & ~(size_t)255;
    return p;
  };
  float* zone = (float*)alloc(BSZ * 4);
  float* hF[2];
  unsigned short* hBF[2];
  for (int p = 0; p < 2; ++p) {
    hF[p]  = (float*)alloc((size_t)3 * HELEM * 4);
    hBF[p] = (unsigned short*)alloc((size_t)3 * HELEM * 2);
  }
  unsigned short* eBF = (unsigned short*)alloc((size_t)BSZ * EW * 2);
  unsigned short* whhF[2][3];
  unsigned short* woF[2][3];
  unsigned short* w1F[2];
  for (int ph = 0; ph < 2; ++ph)
    for (int s = 0; s < 3; ++s) {
      whhF[ph][s] = (unsigned short*)alloc((size_t)768 * 256 * 2);
      woF[ph][s]  = (unsigned short*)alloc((size_t)64 * 256 * 2);
    }
  for (int ph = 0; ph < 2; ++ph)
    w1F[ph] = (unsigned short*)alloc((size_t)256 * 192 * 2);

  // ---- weight swizzle (once per call; deterministic) ----
  for (int ph = 0; ph < 2; ++ph) {
    for (int s = 0; s < 3; ++s) {
      prep_frag<<<768, 256, 0, stream>>>(br[ph][s].Whh, whhF[ph][s], 256,
                                         768 * 256);
      prep_frag<<<64, 256, 0, stream>>>(br[ph][s].W, woF[ph][s], 256,
                                        64 * 256);
    }
    prep_frag<<<192, 256, 0, stream>>>(hd[ph].W1, w1F[ph], 192, 256 * 192);
  }

  k_init<<<768, 256, 0, stream>>>(hF[0], hBF[0], zone, X);

  // ---- per-phase kernarg structs ----
  Branch3 B3[2];
  Out3 O3[2];
  HeadK H2[2];
  const int in_n_[3]  = {5, 4, 1};
  const int uz_[3]    = {1, 1, 0};
  const int nx_[3]    = {4, 3, 1};
  const int fidx_[3][4] = {{1, 2, 3, 4}, {3, 4, 5, 0}, {7, 0, 0, 0}};
  for (int ph = 0; ph < 2; ++ph) {
    for (int s = 0; s < 3; ++s) {
      BranchK& K = B3[ph].b[s];
      K.Wih = br[ph][s].Wih;
      K.bih = br[ph][s].bih;
      K.bhh = br[ph][s].bhh;
      K.whhF = whhF[ph][s];
      K.in_n = in_n_[s];
      K.use_zone = uz_[s];
      K.nx = nx_[s];
      for (int q = 0; q < 4; ++q) K.fidx[q] = fidx_[s][q];
      O3[ph].woF[s] = woF[ph][s];
      O3[ph].bo[s]  = br[ph][s].b;
    }
    H2[ph].w1F = w1F[ph];
    H2[ph].b1  = hd[ph].b1;
    H2[ph].W2  = hd[ph].W2;
  }

  // ---- sequential scan: 3 fused kernels per timestep ----
  float* out = (float*)d_out;
  for (int t = 0; t < TSTP; ++t) {
    int ph = (t < ENCO) ? 0 : 1;
    if (t == ENCO) k_zone_reset<<<1, 256, 0, stream>>>(zone, X);
    int cp = t & 1, npar = cp ^ 1;
    k1_gru<<<48, 256, 0, stream>>>(B3[ph], hBF[cp], hF[cp], hF[npar],
                                   hBF[npar], zone, X, t);
    k2_out<<<12, 256, 0, stream>>>(O3[ph], hBF[npar], eBF);
    k3_head<<<16, 256, 0, stream>>>(H2[ph], eBF, zone, out, t);
  }
}
